// SegGPTAttention_89343909691474
// MI455X (gfx1250) — matrix operand order repaired and checked by reading, not compiled, not run on hardware
//
#include <hip/hip_runtime.h>

// ---------------- problem constants ----------------
#define Bn     4
#define Hn     56
#define Wn     28
#define Cn     1024
#define NHEADS 16
#define HD     64
#define HWn    1568            // 56*28
#define BH     64              // Bn*NHEADS
#define NROWS  6272            // Bn*HWn
#define NQKV   3072

typedef __bf16 bf16;
typedef __attribute__((ext_vector_type(16))) __bf16 v16bf;
typedef __attribute__((ext_vector_type(8)))  __bf16 v8bf;
typedef __attribute__((ext_vector_type(8)))  float  v8f;
typedef __attribute__((ext_vector_type(4)))  float  v4f;
typedef __attribute__((ext_vector_type(4)))  int    v4i;

// ---- optional CDNA5 async global->LDS path (compile-safe guards) ----
#ifndef USE_ASYNC_LDS
#if defined(__has_builtin)
#if __has_builtin(__builtin_amdgcn_global_load_async_to_lds_b128) && \
    __has_builtin(__builtin_amdgcn_s_wait_asynccnt)
#define USE_ASYNC_LDS 1
#endif
#endif
#endif
#ifndef USE_ASYNC_LDS
#define USE_ASYNC_LDS 0
#endif

#if USE_ASYNC_LDS
typedef __attribute__((address_space(1))) v4i* gasync_p;   // global-AS 128-bit vector ptr
typedef __attribute__((address_space(3))) v4i* lasync_p;   // LDS-AS 128-bit vector ptr
__device__ __forceinline__ void async_copy_b128(const void* gsrc, void* lds) {
    __builtin_amdgcn_global_load_async_to_lds_b128(
        (gasync_p)(unsigned long long)gsrc,   // AS1 addr == flat addr for global mem
        (lasync_p)lds, 0, 0);
}
#endif

__device__ __forceinline__ v16bf bf_combine(v8bf lo, v8bf hi) {
    v16bf r;
#pragma unroll
    for (int i = 0; i < 8; ++i) { r[i] = lo[i]; r[i + 8] = hi[i]; }
    return r;
}

// A-fragment (16x32 bf16, MxK): row = lane%16, two contiguous 8-half chunks at
// K = 8*(lane/16) and K = 16 + 8*(lane/16).   (ISA 7.12.2)
__device__ __forceinline__ v16bf load_afrag(const bf16* __restrict__ row, int half) {
    return bf_combine(*(const v8bf*)(row + 8 * half),
                      *(const v8bf*)(row + 16 + 8 * half));
}
// B-fragment (32x16 bf16, KxN): col = lane%16, two 8-half chunks at K = 8*(lane/16) and K = 16 + 8*(lane/16).
__device__ __forceinline__ v16bf load_bfrag(const bf16* __restrict__ col, int half) {
    const bf16* p = col + 8 * half;
    return bf_combine(*(const v8bf*)(p), *(const v8bf*)(p + 16));
}

// ---------------- conversion / transpose kernels ----------------
__global__ void conv_f32_to_bf16(const float* __restrict__ src, bf16* __restrict__ dst, int n) {
    int i = blockIdx.x * blockDim.x + threadIdx.x;
    if (i < n) dst[i] = (bf16)src[i];
}

// W[K][N] fp32 -> Wt[N][K] bf16
__global__ void conv_transpose_bf16(const float* __restrict__ src, bf16* __restrict__ dst,
                                    int K, int N) {
    int i = blockIdx.x * blockDim.x + threadIdx.x;
    if (i >= K * N) return;
    int c = i % K;
    int j = i / K;
    dst[(size_t)j * K + c] = (bf16)src[(size_t)c * N + j];
}

// ---------------- QKV GEMM:  Xb[6272,1024] x Wt[3072,1024]^T ----------------
// One wave computes a 32x64 tile (2 M-subtiles share every B fragment).
__global__ __launch_bounds__(32)
void qkv_gemm(const bf16* __restrict__ Xb, const bf16* __restrict__ Wt,
              const float* __restrict__ qkv_b,
              bf16* __restrict__ Qb, bf16* __restrict__ Kb, bf16* __restrict__ Vb) {
    const int tile = blockIdx.x;
    const int tm = tile / 48, tn = tile % 48;
    const int r0 = tm * 32, col0 = tn * 64;
    const int lane = threadIdx.x & 31;
    const int half = lane >> 4, l16 = lane & 15;

    v8f acc[2][4] = {};
    const bf16* arow0 = Xb + (size_t)(r0 + l16) * Cn;
    const bf16* arow1 = arow0 + (size_t)16 * Cn;
    for (int kb = 0; kb < Cn; kb += 32) {
        __builtin_prefetch(arow0 + kb + 256, 0, 1);
        v16bf A0 = load_afrag(arow0 + kb, half);
        v16bf A1 = load_afrag(arow1 + kb, half);
#pragma unroll
        for (int f = 0; f < 4; ++f) {
            v16bf Bf = load_bfrag(Wt + (size_t)(col0 + f * 16 + l16) * Cn + kb, half);
            acc[0][f] = __builtin_amdgcn_wmma_f32_16x16x32_bf16(
                false, A0, false, Bf, (short)0, acc[0][f], false, false);
            acc[1][f] = __builtin_amdgcn_wmma_f32_16x16x32_bf16(
                false, A1, false, Bf, (short)0, acc[1][f], false, false);
        }
    }
    // epilogue: bias, fold softmax scale into K, scatter into [head][pos][hd]
#pragma unroll
    for (int f = 0; f < 4; ++f) {
        const int j = col0 + f * 16 + l16;
        const int which = j >> 10;            // 0=q 1=k 2=v
        const int head = (j & 1023) >> 6;
        const int c = j & 63;
        const float bias = qkv_b[j];
        bf16* dst = (which == 0) ? Qb : (which == 1) ? Kb : Vb;
        const float sc = (which == 1) ? 0.125f : 1.0f;   // 1/sqrt(64) folded into K
#pragma unroll
        for (int mi = 0; mi < 2; ++mi)
#pragma unroll
            for (int i = 0; i < 8; ++i) {
                const int row = r0 + mi * 16 + i + 8 * half;
                const int b = row / HWn, p = row % HWn;
                const float v = (acc[mi][f][i] + bias) * sc;
                dst[((size_t)(b * NHEADS + head) * HWn + p) * HD + c] = (bf16)v;
            }
    }
}

// ---------------- decomposed rel-pos bias tables ----------------
__global__ void relh_kernel(const bf16* __restrict__ Qb, const float* __restrict__ rel_pos_h,
                            float* __restrict__ Rh) {
    int idx = blockIdx.x * blockDim.x + threadIdx.x;
    if (idx >= BH * HWn * Hn) return;
    const int kh = idx % Hn;
    const int t = idx / Hn;
    const int p = t % HWn;
    const int bh = t / HWn;
    const int h = p / Wn;
    const bf16* qr = Qb + ((size_t)bh * HWn + p) * HD;
    const float* rr = rel_pos_h + (size_t)(h - kh + Hn - 1) * HD;
    float s = 0.f;
#pragma unroll 8
    for (int c = 0; c < HD; ++c) s += (float)qr[c] * rr[c];
    Rh[idx] = s;
}

__global__ void relw_kernel(const bf16* __restrict__ Qb, const float* __restrict__ rel_pos_w,
                            float* __restrict__ Rw) {
    int idx = blockIdx.x * blockDim.x + threadIdx.x;
    if (idx >= BH * HWn * Wn) return;
    const int kw = idx % Wn;
    const int t = idx / Wn;
    const int p = t % HWn;
    const int bh = t / HWn;
    const int w = p % Wn;
    const bf16* qr = Qb + ((size_t)bh * HWn + p) * HD;
    const float* rr = rel_pos_w + (size_t)(w - kw + Wn - 1) * HD;
    float s = 0.f;
#pragma unroll 8
    for (int c = 0; c < HD; ++c) s += (float)qr[c] * rr[c];
    Rw[idx] = s;
}

// ---------------- flash attention with decomposed bias ----------------
// Block = 64 threads (2 waves), one (head, 32-query) tile; 49 key steps of 32.
#define LOG2E 1.44269504088896f
__global__ __launch_bounds__(64)
void attn_kernel(const bf16* __restrict__ Qb, const bf16* __restrict__ Kb,
                 const bf16* __restrict__ Vb, const float* __restrict__ Rh,
                 const float* __restrict__ Rw, bf16* __restrict__ X2) {
    __shared__ __align__(16) bf16 Kt[32 * HD];        // [key][c]
    __shared__ __align__(16) bf16 Vt[HD * 32];        // [c][key]
    __shared__ __align__(16) bf16 Pls[2][16 * 32];    // per-wave P scratch
    __shared__ __align__(16) float Bh[Hn * 32];       // bias slice [kh][qlocal]
    __shared__ __align__(16) float Bw[Wn * 32];       // bias slice [kw][qlocal]

    const int bh = blockIdx.x / 49;
    const int q0 = (blockIdx.x % 49) * 32;
    const int tid = threadIdx.x;
    const int wave = tid >> 5;
    const int lane = tid & 31;
    const int half = lane >> 4, l16 = lane & 15;
    const int qrow = q0 + wave * 16;

    const bf16* qhead = Qb + (size_t)bh * HWn * HD;
    const bf16* khead = Kb + (size_t)bh * HWn * HD;
    const bf16* vhead = Vb + (size_t)bh * HWn * HD;
    const float* rhh = Rh + (size_t)bh * HWn * Hn;
    const float* rwh = Rw + (size_t)bh * HWn * Wn;

    // stage this block's bias slices into LDS, transposed to [k][qlocal]
    for (int idx = tid; idx < Hn * 32; idx += 64) {
        const int kh = idx >> 5, ql = idx & 31;
        Bh[idx] = rhh[(size_t)(q0 + ql) * Hn + kh];
    }
    for (int idx = tid; idx < Wn * 32; idx += 64) {
        const int kw = idx >> 5, ql = idx & 31;
        Bw[idx] = rwh[(size_t)(q0 + ql) * Wn + kw];
    }

    // Q A-fragments, resident for the whole K loop
    v16bf Aq[2];
    {
        const bf16* qr = qhead + (size_t)(qrow + l16) * HD;
#pragma unroll
        for (int cb = 0; cb < 2; ++cb) Aq[cb] = load_afrag(qr + cb * 32, half);
    }

    float m[8], l[8];
#pragma unroll
    for (int i = 0; i < 8; ++i) { m[i] = -1e30f; l[i] = 0.f; }
    v8f O[4] = {};

    for (int kt = 0; kt < HWn; kt += 32) {
        __syncthreads();                       // previous tile fully consumed
        // ---- K tile -> LDS (async path when available) ----
        {
            const bf16* src = khead + (size_t)kt * HD;
#if USE_ASYNC_LDS
#pragma unroll
            for (int u = 0; u < 4; ++u) {
                const int e = tid * 32 + u * 8;
                async_copy_b128(src + e, Kt + e);
            }
#else
#pragma unroll
            for (int u = 0; u < 4; ++u) {
                const int e = tid * 32 + u * 8;
                *(v8bf*)(Kt + e) = *(const v8bf*)(src + e);
            }
#endif
        }
        // ---- V tile -> LDS transposed (coalesced v8bf reads, scalar ds writes) ----
        {
            const bf16* vsrc = vhead + (size_t)kt * HD;
            const int key = tid >> 1;          // 0..31
            const int c0 = (tid & 1) * 32;     // 0 or 32
            v8bf r[4];
#pragma unroll
            for (int u = 0; u < 4; ++u)
                r[u] = *(const v8bf*)(vsrc + (size_t)key * HD + c0 + u * 8);
#pragma unroll
            for (int u = 0; u < 4; ++u)
#pragma unroll
                for (int j = 0; j < 8; ++j)
                    Vt[(c0 + u * 8 + j) * 32 + key] = r[u][j];
        }
#if USE_ASYNC_LDS
        __builtin_amdgcn_s_wait_asynccnt(0);
#endif
        __syncthreads();

        // S = Q * K^T  (16x32 = two 16x16 D-frags, each sum of 2 WMMAs)
        v8f S[2] = {};
#pragma unroll
        for (int ks = 0; ks < 2; ++ks)
#pragma unroll
            for (int cb = 0; cb < 2; ++cb) {
                v16bf Bf = load_bfrag(Kt + (ks * 16 + l16) * HD + cb * 32, half);
                S[ks] = __builtin_amdgcn_wmma_f32_16x16x32_bf16(
                    false, Aq[cb], false, Bf, (short)0, S[ks], false, false);
            }

        // decomposed rel-pos bias from the LDS slices (vectorized v4f reads)
#pragma unroll
        for (int ks = 0; ks < 2; ++ks) {
            const int kpos = kt + ks * 16 + l16;
            const int kh = kpos / Wn, kw = kpos % Wn;
            const int qb = wave * 16 + 8 * half;
            const v4f h0 = *(const v4f*)(Bh + kh * 32 + qb);
            const v4f h1 = *(const v4f*)(Bh + kh * 32 + qb + 4);
            const v4f w0 = *(const v4f*)(Bw + kw * 32 + qb);
            const v4f w1 = *(const v4f*)(Bw + kw * 32 + qb + 4);
#pragma unroll
            for (int i = 0; i < 4; ++i) {
                S[ks][i]     += h0[i] + w0[i];
                S[ks][i + 4] += h1[i] + w1[i];
            }
        }

        // online softmax (each query row lives in one 16-lane half)
        float alpha[8];
#pragma unroll
        for (int i = 0; i < 8; ++i) {
            float rm = fmaxf(S[0][i], S[1][i]);
            rm = fmaxf(rm, __shfl_xor(rm, 1, 32));
            rm = fmaxf(rm, __shfl_xor(rm, 2, 32));
            rm = fmaxf(rm, __shfl_xor(rm, 4, 32));
            rm = fmaxf(rm, __shfl_xor(rm, 8, 32));
            const float mn = fmaxf(m[i], rm);
            alpha[i] = exp2f((m[i] - mn) * LOG2E);
            m[i] = mn;
            const float p0 = exp2f((S[0][i] - mn) * LOG2E);
            const float p1 = exp2f((S[1][i] - mn) * LOG2E);
            S[0][i] = p0; S[1][i] = p1;
            float rs = p0 + p1;
            rs += __shfl_xor(rs, 1, 32);
            rs += __shfl_xor(rs, 2, 32);
            rs += __shfl_xor(rs, 4, 32);
            rs += __shfl_xor(rs, 8, 32);
            l[i] = l[i] * alpha[i] + rs;
        }
#pragma unroll
        for (int f = 0; f < 4; ++f)
#pragma unroll
            for (int i = 0; i < 8; ++i) O[f][i] *= alpha[i];

        // P (f32, D-layout) -> bf16 LDS -> A-layout fragment
        bf16* P = Pls[wave];
#pragma unroll
        for (int ks = 0; ks < 2; ++ks)
#pragma unroll
            for (int i = 0; i < 8; ++i)
                P[(i + 8 * half) * 32 + ks * 16 + l16] = (bf16)S[ks][i];
        __syncthreads();

        v16bf Ap = load_afrag(P + l16 * 32, half);
#pragma unroll
        for (int f = 0; f < 4; ++f) {
            v16bf Bf = load_bfrag(Vt + (f * 16 + l16) * 32, half);
            O[f] = __builtin_amdgcn_wmma_f32_16x16x32_bf16(
                false, Ap, false, Bf, (short)0, O[f], false, false);
        }
    }

    // normalize and scatter into [b, p, head*64 + c] bf16 for the proj GEMM
    const int b = bh >> 4, head = bh & 15;
#pragma unroll
    for (int i = 0; i < 8; ++i) {
        const float inv = 1.f / l[i];
        const int qp = qrow + i + 8 * half;
        const size_t row = (size_t)(b * HWn + qp) * Cn + head * HD;
#pragma unroll
        for (int f = 0; f < 4; ++f)
            X2[row + f * 16 + l16] = (bf16)(O[f][i] * inv);
    }
}

// ---------------- output projection: X2[6272,1024] x Pt[1024,1024]^T ----------------
__global__ __launch_bounds__(32)
void proj_gemm(const bf16* __restrict__ X2, const bf16* __restrict__ Pt,
               const float* __restrict__ proj_b, float* __restrict__ out) {
    const int tile = blockIdx.x;
    const int tm = tile >> 4, tn = tile & 15;
    const int r0 = tm * 32, col0 = tn * 64;
    const int lane = threadIdx.x & 31;
    const int half = lane >> 4, l16 = lane & 15;

    v8f acc[2][4] = {};
    const bf16* arow0 = X2 + (size_t)(r0 + l16) * Cn;
    const bf16* arow1 = arow0 + (size_t)16 * Cn;
    for (int kb = 0; kb < Cn; kb += 32) {
        __builtin_prefetch(arow0 + kb + 256, 0, 1);
        v16bf A0 = load_afrag(arow0 + kb, half);
        v16bf A1 = load_afrag(arow1 + kb, half);
#pragma unroll
        for (int f = 0; f < 4; ++f) {
            v16bf Bf = load_bfrag(Pt + (size_t)(col0 + f * 16 + l16) * Cn + kb, half);
            acc[0][f] = __builtin_amdgcn_wmma_f32_16x16x32_bf16(
                false, A0, false, Bf, (short)0, acc[0][f], false, false);
            acc[1][f] = __builtin_amdgcn_wmma_f32_16x16x32_bf16(
                false, A1, false, Bf, (short)0, acc[1][f], false, false);
        }
    }
#pragma unroll
    for (int f = 0; f < 4; ++f) {
        const int j = col0 + f * 16 + l16;
        const float bias = proj_b[j];
#pragma unroll
        for (int mi = 0; mi < 2; ++mi)
#pragma unroll
            for (int i = 0; i < 8; ++i) {
                const int row = r0 + mi * 16 + i + 8 * half;
                out[(size_t)row * Cn + j] = acc[mi][f][i] + bias;
            }
    }
}

// ---------------- host launch ----------------
extern "C" void kernel_launch(void* const* d_in, const int* in_sizes, int n_in,
                              void* d_out, int out_size, void* d_ws, size_t ws_size,
                              hipStream_t stream) {
    (void)in_sizes; (void)n_in; (void)out_size; (void)ws_size;
    const float* hidden    = (const float*)d_in[0];
    const float* qkv_w     = (const float*)d_in[1];
    const float* qkv_b     = (const float*)d_in[2];
    const float* proj_w    = (const float*)d_in[3];
    const float* proj_b    = (const float*)d_in[4];
    const float* rel_pos_h = (const float*)d_in[5];
    const float* rel_pos_w = (const float*)d_in[6];
    float* out = (float*)d_out;

    char* ws = (char*)d_ws;
    size_t off = 0;
    auto carve = [&](size_t bytes) {
        char* p = ws + off;
        off += (bytes + 255) & ~(size_t)255;
        return p;
    };
    bf16* Xb  = (bf16*)carve((size_t)NROWS * Cn * 2);
    bf16* Wt  = (bf16*)carve((size_t)NQKV * Cn * 2);
    bf16* Pt  = (bf16*)carve((size_t)Cn * Cn * 2);
    bf16* Qb  = (bf16*)carve((size_t)BH * HWn * HD * 2);
    bf16* Kb  = (bf16*)carve((size_t)BH * HWn * HD * 2);
    bf16* Vb  = (bf16*)carve((size_t)BH * HWn * HD * 2);
    float* Rh = (float*)carve((size_t)BH * HWn * Hn * 4);
    float* Rw = (float*)carve((size_t)BH * HWn * Wn * 4);
    bf16* X2  = (bf16*)carve((size_t)NROWS * Cn * 2);

    const int nX = NROWS * Cn;
    conv_f32_to_bf16<<<(nX + 255) / 256, 256, 0, stream>>>(hidden, Xb, nX);
    conv_transpose_bf16<<<(Cn * NQKV + 255) / 256, 256, 0, stream>>>(qkv_w, Wt, Cn, NQKV);
    conv_transpose_bf16<<<(Cn * Cn + 255) / 256, 256, 0, stream>>>(proj_w, Pt, Cn, Cn);

    qkv_gemm<<<(NROWS / 32) * (NQKV / 64), 32, 0, stream>>>(Xb, Wt, qkv_b, Qb, Kb, Vb);

    const int nRh = BH * HWn * Hn;
    const int nRw = BH * HWn * Wn;
    relh_kernel<<<(nRh + 255) / 256, 256, 0, stream>>>(Qb, rel_pos_h, Rh);
    relw_kernel<<<(nRw + 255) / 256, 256, 0, stream>>>(Qb, rel_pos_w, Rw);

    attn_kernel<<<BH * (HWn / 32), 64, 0, stream>>>(Qb, Kb, Vb, Rh, Rw, X2);

    proj_gemm<<<(NROWS / 32) * (Cn / 64), 32, 0, stream>>>(X2, Pt, proj_b, out);
}
